// MoE_82678120448637
// MI455X (gfx1250) — compile-verified, matrix operand order checked by reading, not run on hardware
//
#include <hip/hip_runtime.h>
#include <hip/hip_bf16.h>
#include <stdint.h>

// Problem constants (match reference setup_inputs)
#define NTOK 16384
#define DD   1024
#define HH   512
#define NE   64
#define CAP  1024
#define TOPK 2

typedef __attribute__((ext_vector_type(16))) __bf16 v16bf;
typedef __attribute__((ext_vector_type(8)))  float  v8f;

union Frag16 { uint4 q[2]; v16bf v; };

static __device__ __forceinline__ unsigned short f32_to_bf16_rne(float f) {
    unsigned int u = __float_as_uint(f);
    u += 0x7FFFu + ((u >> 16) & 1u);   // round-to-nearest-even
    return (unsigned short)(u >> 16);
}

static __device__ __forceinline__ void load_frag(Frag16& f, const unsigned short* p) {
    f.q[0] = *(const uint4*)(p);
    f.q[1] = *(const uint4*)(p + 16);
}

// ---------------- init: counts = 0, slot_token = -1 ----------------
__global__ void moe_init_kernel(int* __restrict__ counts, int* __restrict__ slot_token) {
    int i = blockIdx.x * blockDim.x + threadIdx.x;
    if (i < NE) counts[i] = 0;
    if (i < NE * CAP) slot_token[i] = -1;
}

// ---------------- fp32 -> bf16 convert ----------------
__global__ __launch_bounds__(256)
void moe_cvt_kernel(const float* __restrict__ src, unsigned short* __restrict__ dst, long n) {
    long i = (long)blockIdx.x * blockDim.x + threadIdx.x;
    long stride = (long)gridDim.x * blockDim.x;
    for (; i < n; i += stride) dst[i] = f32_to_bf16_rne(src[i]);
}

// ---------------- router: sigmoid(x@gate_w^T), bias-assisted top-2 ----------------
__global__ __launch_bounds__(64)
void moe_router_kernel(const float* __restrict__ x, const float* __restrict__ gate_w,
                       const float* __restrict__ ebias,
                       int* __restrict__ idx2, float* __restrict__ prob2) {
    __shared__ float xs[DD];
    __shared__ float sc[NE];
    int token = blockIdx.x;
    const float* xr = x + (long)token * DD;
    for (int d = threadIdx.x; d < DD; d += 64) xs[d] = xr[d];
    __syncthreads();
    int e = threadIdx.x;
    const float* gw = gate_w + (long)e * DD;
    float acc = 0.f;
    #pragma unroll 8
    for (int d = 0; d < DD; ++d) acc += xs[d] * gw[d];
    sc[e] = 1.0f / (1.0f + __expf(-acc));
    __syncthreads();
    if (threadIdx.x == 0) {
        float b0 = -1e30f, b1 = -1e30f; int i0 = 0, i1 = 0;
        for (int j = 0; j < NE; ++j) {
            float v = sc[j] + ebias[j];
            if (v > b0) { b1 = b0; i1 = i0; b0 = v; i0 = j; }
            else if (v > b1) { b1 = v; i1 = j; }
        }
        float s0 = sc[i0], s1 = sc[i1];
        float inv = 1.0f / (s0 + s1 + 1e-20f);
        idx2[token * 2 + 0] = i0; idx2[token * 2 + 1] = i1;
        prob2[token * 2 + 0] = s0 * inv; prob2[token * 2 + 1] = s1 * inv;
    }
}

// ---------------- assign: rank within expert -> slot ----------------
__global__ __launch_bounds__(256)
void moe_assign_kernel(const int* __restrict__ idx2, const float* __restrict__ prob2,
                       int* __restrict__ counts,
                       int* __restrict__ slot_token, float* __restrict__ slot_prob) {
    int i = blockIdx.x * blockDim.x + threadIdx.x;   // over N*K token-copies
    if (i >= NTOK * TOPK) return;
    int e = idx2[i];
    int pos = atomicAdd(&counts[e], 1);
    if (pos < CAP) {
        int slot = e * CAP + pos;
        slot_token[slot] = i >> 1;
        slot_prob[slot]  = prob2[i];
    }
}

// ---------------- dispatch: xb[slot] = bf16(x[token] * prob), zeros if empty ----------------
__global__ __launch_bounds__(256)
void moe_dispatch_kernel(const float* __restrict__ x, const int* __restrict__ slot_token,
                         const float* __restrict__ slot_prob, unsigned short* __restrict__ xb) {
    int slot = blockIdx.x;
    int t = slot_token[slot];
    float p = (t >= 0) ? slot_prob[slot] : 0.f;
    const float* xr = x + (long)((t >= 0) ? t : 0) * DD;
    unsigned short* o = xb + (long)slot * DD;
    for (int d = threadIdx.x; d < DD; d += 256) {
        float v = (t >= 0) ? xr[d] * p : 0.f;
        o[d] = f32_to_bf16_rne(v);
    }
}

// ---- WMMA SwiGLU GEMM, 32x32 tile per wave (2x2 register blocking) ----
// Hout = bf16( silu(A@B1^T) * (A@B3^T) ); A:[M,K], B1/B3:[N,K] row-major bf16.
__global__ __launch_bounds__(256)
void moe_gemm_swiglu_kernel(const unsigned short* __restrict__ A,
                            const unsigned short* __restrict__ B1,
                            const unsigned short* __restrict__ B3,
                            unsigned short* __restrict__ Hout,
                            int M, int N, int K,
                            long strideA, long strideB, long strideH) {
    int e = blockIdx.y;
    const unsigned short* Ae  = A  + (long)e * strideA;
    const unsigned short* B1e = B1 + (long)e * strideB;
    const unsigned short* B3e = B3 + (long)e * strideB;
    unsigned short* He = Hout + (long)e * strideH;

    int lane = threadIdx.x & 31;
    int wave = threadIdx.x >> 5;
    int tilesN = N >> 5, tilesM = M >> 5;          // 32x32 wave tiles
    int tile = blockIdx.x * 8 + wave;              // wave-uniform
    if (tile >= tilesM * tilesN) return;           // whole-wave exit, EXEC stays all-1
    int tm = tile / tilesN, tn = tile % tilesN;
    int m0 = tm << 5, n0 = tn << 5;

    int rlane = lane & 15;
    int khalf = (lane >> 4) << 3;                  // 0 or 8

    const unsigned short* ap0  = Ae  + (long)(m0 + rlane)      * K + khalf;
    const unsigned short* ap1  = Ae  + (long)(m0 + 16 + rlane) * K + khalf;
    const unsigned short* b1p0 = B1e + (long)(n0 + rlane)      * K + khalf;
    const unsigned short* b1p1 = B1e + (long)(n0 + 16 + rlane) * K + khalf;
    const unsigned short* b3p0 = B3e + (long)(n0 + rlane)      * K + khalf;
    const unsigned short* b3p1 = B3e + (long)(n0 + 16 + rlane) * K + khalf;

    v8f a1_00 = {}, a1_01 = {}, a1_10 = {}, a1_11 = {};
    v8f a3_00 = {}, a3_01 = {}, a3_10 = {}, a3_11 = {};
    for (int kk = 0; kk < K; kk += 32) {
        Frag16 a0, a1, b10, b11, b30, b31;
        load_frag(a0,  ap0  + kk);  load_frag(a1,  ap1  + kk);
        load_frag(b10, b1p0 + kk);  load_frag(b11, b1p1 + kk);
        load_frag(b30, b3p0 + kk);  load_frag(b31, b3p1 + kk);
        a1_00 = __builtin_amdgcn_wmma_f32_16x16x32_bf16(false, a0.v, false, b10.v, (short)0, a1_00, false, false);
        a1_01 = __builtin_amdgcn_wmma_f32_16x16x32_bf16(false, a0.v, false, b11.v, (short)0, a1_01, false, false);
        a1_10 = __builtin_amdgcn_wmma_f32_16x16x32_bf16(false, a1.v, false, b10.v, (short)0, a1_10, false, false);
        a1_11 = __builtin_amdgcn_wmma_f32_16x16x32_bf16(false, a1.v, false, b11.v, (short)0, a1_11, false, false);
        a3_00 = __builtin_amdgcn_wmma_f32_16x16x32_bf16(false, a0.v, false, b30.v, (short)0, a3_00, false, false);
        a3_01 = __builtin_amdgcn_wmma_f32_16x16x32_bf16(false, a0.v, false, b31.v, (short)0, a3_01, false, false);
        a3_10 = __builtin_amdgcn_wmma_f32_16x16x32_bf16(false, a1.v, false, b30.v, (short)0, a3_10, false, false);
        a3_11 = __builtin_amdgcn_wmma_f32_16x16x32_bf16(false, a1.v, false, b31.v, (short)0, a3_11, false, false);
    }

    int rbase = m0 + ((lane >> 4) << 3);
    int cbase = n0 + rlane;
    v8f* g1[4] = { &a1_00, &a1_01, &a1_10, &a1_11 };
    v8f* g3[4] = { &a3_00, &a3_01, &a3_10, &a3_11 };
    #pragma unroll
    for (int t = 0; t < 4; ++t) {
        int orow0 = rbase + ((t >> 1) << 4);
        int ocol  = cbase + ((t & 1) << 4);
        #pragma unroll
        for (int v = 0; v < 8; ++v) {
            float z = (*g1[t])[v];
            float s = z / (1.0f + __expf(-z));     // silu
            He[(long)(orow0 + v) * N + ocol] = f32_to_bf16_rne(s * (*g3[t])[v]);
        }
    }
}

// ---- WMMA down-proj GEMM, 32x32 tile per wave (2x2 register blocking) ----
// mode 0: plain store Out[row*N+col] (shared expert; also initializes d_out)
// mode 1: atomicAdd into Out[token*N+col] via slot_token (routed combine)
__global__ __launch_bounds__(256)
void moe_gemm_out_kernel(const unsigned short* __restrict__ A,
                         const unsigned short* __restrict__ B,
                         float* __restrict__ Out,
                         const int* __restrict__ slot_token,
                         int M, int N, int K,
                         long strideA, long strideB, int mode) {
    int e = blockIdx.y;
    const unsigned short* Ae = A + (long)e * strideA;
    const unsigned short* Be = B + (long)e * strideB;

    int lane = threadIdx.x & 31;
    int wave = threadIdx.x >> 5;
    int tilesN = N >> 5, tilesM = M >> 5;
    int tile = blockIdx.x * 8 + wave;
    if (tile >= tilesM * tilesN) return;
    int tm = tile / tilesN, tn = tile % tilesN;
    int m0 = tm << 5, n0 = tn << 5;

    int rlane = lane & 15;
    int khalf = (lane >> 4) << 3;

    const unsigned short* ap0 = Ae + (long)(m0 + rlane)      * K + khalf;
    const unsigned short* ap1 = Ae + (long)(m0 + 16 + rlane) * K + khalf;
    const unsigned short* bp0 = Be + (long)(n0 + rlane)      * K + khalf;
    const unsigned short* bp1 = Be + (long)(n0 + 16 + rlane) * K + khalf;

    v8f c00 = {}, c01 = {}, c10 = {}, c11 = {};
    for (int kk = 0; kk < K; kk += 32) {
        Frag16 a0, a1, b0, b1;
        load_frag(a0, ap0 + kk);  load_frag(a1, ap1 + kk);
        load_frag(b0, bp0 + kk);  load_frag(b1, bp1 + kk);
        c00 = __builtin_amdgcn_wmma_f32_16x16x32_bf16(false, a0.v, false, b0.v, (short)0, c00, false, false);
        c01 = __builtin_amdgcn_wmma_f32_16x16x32_bf16(false, a0.v, false, b1.v, (short)0, c01, false, false);
        c10 = __builtin_amdgcn_wmma_f32_16x16x32_bf16(false, a1.v, false, b0.v, (short)0, c10, false, false);
        c11 = __builtin_amdgcn_wmma_f32_16x16x32_bf16(false, a1.v, false, b1.v, (short)0, c11, false, false);
    }

    int rbase = m0 + ((lane >> 4) << 3);
    int cbase = n0 + rlane;
    v8f* g[4] = { &c00, &c01, &c10, &c11 };
    #pragma unroll
    for (int t = 0; t < 4; ++t) {
        int orow0 = rbase + ((t >> 1) << 4);
        int ocol  = cbase + ((t & 1) << 4);
        if (mode == 0) {
            #pragma unroll
            for (int v = 0; v < 8; ++v)
                Out[(long)(orow0 + v) * N + ocol] = (*g[t])[v];
        } else {
            #pragma unroll
            for (int v = 0; v < 8; ++v) {
                int token = slot_token[(long)e * M + orow0 + v];
                if (token >= 0)
                    atomicAdd(&Out[(long)token * N + ocol], (*g[t])[v]);
            }
        }
    }
}

// ---------------- host launcher ----------------
static inline size_t align256(size_t x) { return (x + 255) & ~(size_t)255; }

extern "C" void kernel_launch(void* const* d_in, const int* in_sizes, int n_in,
                              void* d_out, int out_size, void* d_ws, size_t ws_size,
                              hipStream_t stream) {
    const float* x      = (const float*)d_in[0];  // [N,D]
    const float* gate_w = (const float*)d_in[1];  // [E,D]
    const float* w1     = (const float*)d_in[2];  // [E,H,D]
    const float* w2     = (const float*)d_in[3];  // [E,D,H]
    const float* w3     = (const float*)d_in[4];  // [E,H,D]
    const float* w1s    = (const float*)d_in[5];  // [H,D]
    const float* w2s    = (const float*)d_in[6];  // [D,H]
    const float* w3s    = (const float*)d_in[7];  // [H,D]
    const float* ebias  = (const float*)d_in[8];  // [E]
    float* out = (float*)d_out;                   // [N,D]

    // ---- workspace carve (bf16 stored as ushort) ----
    char* ws = (char*)d_ws;
    size_t off = 0;
    auto take = [&](size_t bytes) { char* p = ws + off; off = align256(off + bytes); return p; };
    unsigned short* xbf   = (unsigned short*)take((size_t)NTOK * DD * 2);
    unsigned short* w1b   = (unsigned short*)take((size_t)NE * HH * DD * 2);
    unsigned short* w3b   = (unsigned short*)take((size_t)NE * HH * DD * 2);
    unsigned short* w2b   = (unsigned short*)take((size_t)NE * DD * HH * 2);
    unsigned short* w1sb  = (unsigned short*)take((size_t)HH * DD * 2);
    unsigned short* w3sb  = (unsigned short*)take((size_t)HH * DD * 2);
    unsigned short* w2sb  = (unsigned short*)take((size_t)DD * HH * 2);
    unsigned short* xb    = (unsigned short*)take((size_t)NE * CAP * DD * 2);
    unsigned short* hbuf  = (unsigned short*)take((size_t)NE * CAP * HH * 2);
    unsigned short* hs    = (unsigned short*)take((size_t)NTOK * HH * 2);
    int*   idx2       = (int*)take((size_t)NTOK * TOPK * 4);
    float* prob2      = (float*)take((size_t)NTOK * TOPK * 4);
    int*   counts     = (int*)take((size_t)NE * 4);
    int*   slot_token = (int*)take((size_t)NE * CAP * 4);
    float* slot_prob  = (float*)take((size_t)NE * CAP * 4);
    (void)ws_size; (void)in_sizes; (void)n_in; (void)out_size;

    // 1) init
    moe_init_kernel<<<(NE * CAP + 255) / 256, 256, 0, stream>>>(counts, slot_token);

    // 2) fp32 -> bf16 conversions
    moe_cvt_kernel<<<2048, 256, 0, stream>>>(x,   xbf,  (long)NTOK * DD);
    moe_cvt_kernel<<<2048, 256, 0, stream>>>(w1,  w1b,  (long)NE * HH * DD);
    moe_cvt_kernel<<<2048, 256, 0, stream>>>(w3,  w3b,  (long)NE * HH * DD);
    moe_cvt_kernel<<<2048, 256, 0, stream>>>(w2,  w2b,  (long)NE * DD * HH);
    moe_cvt_kernel<<<512,  256, 0, stream>>>(w1s, w1sb, (long)HH * DD);
    moe_cvt_kernel<<<512,  256, 0, stream>>>(w3s, w3sb, (long)HH * DD);
    moe_cvt_kernel<<<512,  256, 0, stream>>>(w2s, w2sb, (long)DD * HH);

    // 3) router + assign + dispatch
    moe_router_kernel<<<NTOK, 64, 0, stream>>>(x, gate_w, ebias, idx2, prob2);
    moe_assign_kernel<<<(NTOK * TOPK + 255) / 256, 256, 0, stream>>>(idx2, prob2, counts,
                                                                     slot_token, slot_prob);
    moe_dispatch_kernel<<<NE * CAP, 256, 0, stream>>>(x, slot_token, slot_prob, xb);

    // 4) SwiGLU up-projections (WMMA bf16, 32x32 wave tiles)
    {   // experts: per e, [CAP,DD] x [HH,DD]^T -> [CAP,HH]
        dim3 grid((CAP / 32) * (HH / 32) / 8, NE);
        moe_gemm_swiglu_kernel<<<grid, 256, 0, stream>>>(xb, w1b, w3b, hbuf,
            CAP, HH, DD, (long)CAP * DD, (long)HH * DD, (long)CAP * HH);
    }
    {   // shared: [NTOK,DD] x [HH,DD]^T -> [NTOK,HH]
        dim3 grid((NTOK / 32) * (HH / 32) / 8, 1);
        moe_gemm_swiglu_kernel<<<grid, 256, 0, stream>>>(xbf, w1sb, w3sb, hs,
            NTOK, HH, DD, 0L, 0L, 0L);
    }

    // 5) down-projections: shared writes d_out (init), experts atomically combine
    {   // shared: [NTOK,HH] x [DD,HH]^T -> d_out (plain store)
        dim3 grid((NTOK / 32) * (DD / 32) / 8, 1);
        moe_gemm_out_kernel<<<grid, 256, 0, stream>>>(hs, w2sb, out, nullptr,
            NTOK, DD, HH, 0L, 0L, 0);
    }
    {   // experts: per e, [CAP,HH] x [DD,HH]^T -> atomicAdd d_out[token]
        dim3 grid((CAP / 32) * (DD / 32) / 8, NE);
        moe_gemm_out_kernel<<<grid, 256, 0, stream>>>(hbuf, w2b, out, slot_token,
            CAP, DD, HH, (long)CAP * HH, (long)DD * HH, 1);
    }
}
